// DIONPQGO_40123584479354
// MI455X (gfx1250) — matrix operand order repaired and checked by reading, not compiled, not run on hardware
//
#include <hip/hip_runtime.h>
#include <hip/hip_bf16.h>
#include <math.h>

typedef __bf16 bf16;
typedef __attribute__((ext_vector_type(16))) __bf16 v16bf;
typedef __attribute__((ext_vector_type(8)))  __bf16 v8bf;
typedef __attribute__((ext_vector_type(8)))  float  v8f;

#define DD   512
#define KK   2048
#define NN   16384          // 16*32*32
#define BB   16
#define HH   32
#define WW   32
#define BETA 0.25f

// flat float offsets into d_out (return-order concatenation)
#define ZQ_OFF    0ull
#define LOSS_OFF  8388608ull
#define PROB_OFF  8388609ull
#define PPROB_OFF 41943041ull
#define IDX_OFF   75497473ull

// ---------------------------------------------------------------------------
// 1) codebook f32 -> bf16, plus cnorm[k] = ||c_k||^2. One block per code row.
// ---------------------------------------------------------------------------
__global__ __launch_bounds__(256) void prep_codebook_kernel(
    const float* __restrict__ cb, bf16* __restrict__ cb16,
    float* __restrict__ cnorm)
{
    const int k = blockIdx.x, tid = threadIdx.x;
    float s = 0.0f;
    for (int d = tid; d < DD; d += 256) {
        float v = cb[(size_t)k * DD + d];
        cb16[(size_t)k * DD + d] = (bf16)v;
        s += v * v;
    }
    __shared__ float rs[256];
    rs[tid] = s; __syncthreads();
    for (int off = 128; off > 0; off >>= 1) {
        if (tid < off) rs[tid] += rs[tid + off];
        __syncthreads();
    }
    if (tid == 0) cnorm[k] = rs[0];
}

// ---------------------------------------------------------------------------
// 2) z (b,d,h,w) f32 -> A (N,d) bf16, N row = (b,h,w). LDS 32x32 transpose
//    so both the global reads (over w) and writes (over d) are coalesced.
//    One block per (b,h).
// ---------------------------------------------------------------------------
__global__ __launch_bounds__(256) void prep_z_kernel(
    const float* __restrict__ src, bf16* __restrict__ dst)
{
    const int bid = blockIdx.x;
    const int b = bid >> 5, h = bid & 31;
    const int tid = threadIdx.x;
    __shared__ float tile[32][33];
    for (int dt = 0; dt < DD; dt += 32) {
        #pragma unroll
        for (int i = 0; i < 4; ++i) {
            int dl = (tid >> 5) + i * 8;
            int w  = tid & 31;
            tile[dl][w] = src[(((size_t)b * DD + dt + dl) * HH + h) * WW + w];
        }
        __syncthreads();
        #pragma unroll
        for (int i = 0; i < 4; ++i) {
            int wl = (tid >> 5) + i * 8;
            int d2 = tid & 31;
            size_t n = (size_t)b * (HH * WW) + h * WW + wl;
            dst[n * DD + dt + d2] = (bf16)tile[d2][wl];
        }
        __syncthreads();
    }
}

// ---------------------------------------------------------------------------
// 3) Main: WMMA bf16 GEMM tile (16 rows x 2048 cols, k=512), dist in LDS,
//    argmin + numerically stable softmax(-dist), coalesced prob writes.
//    One block (8 waves) per 16 rows.  dist[m][k] = cnorm[k] - 2*dot
//    (the ||x||^2 row-constant cancels in softmax and argmin).
// ---------------------------------------------------------------------------
__global__ __launch_bounds__(256) void vq_main_kernel(
    const bf16*  __restrict__ A,  const bf16* __restrict__ CB,
    const float* __restrict__ cnorm, float* __restrict__ prob_out,
    int* __restrict__ idx_i, float* __restrict__ idx_f)
{
    __shared__ bf16  A_s[16 * DD];       // 16 KB
    __shared__ float dist_s[16 * KK];    // 128 KB
    __shared__ float red_val[256];
    __shared__ int   red_idx[256];
    __shared__ float row_min[16];
    __shared__ float row_inv[16];

    const int tid  = threadIdx.x;
    const int lane = tid & 31;
    const int wv   = tid >> 5;
    const int n0   = blockIdx.x * 16;

    // A tile: 16 consecutive (N,d) rows = one contiguous 16 KB chunk
    {
        const uint4* s = (const uint4*)(A + (size_t)n0 * DD);
        uint4* d = (uint4*)A_s;
        for (int i = tid; i < (16 * DD * 2) / 16; i += 256) d[i] = s[i];
    }
    __syncthreads();

    // ISA 16-bit A layout: lane L holds row m=L&15; for k-step kk the lane's
    // 16 elems are [kk+8g .. kk+8g+7] ++ [kk+16+8g .. kk+16+8g+7], g=L>>4.
// B layout (same k order as A): lane L holds col n=L&15, same k elems as A.
    const int m = lane & 15;
    const int g = lane >> 4;
    const v8bf* arow = (const v8bf*)(A_s + m * DD);

    for (int ct = wv; ct < KK / 16; ct += 8) {
        const int col = ct * 16 + m;
        const bf16* brow = CB + (size_t)col * DD;
        v8f acc = {};
        for (int kk = 0; kk < DD; kk += 32) {
            v8bf lo = arow[(kk >> 3) + g];
            v8bf hi = arow[(kk >> 3) + 2 + g];
            v16bf afrag = __builtin_shufflevector(
                lo, hi, 0,1,2,3,4,5,6,7,8,9,10,11,12,13,14,15);
            v16bf bfrag = __builtin_shufflevector(
                *(const v8bf*)(brow + kk + 8 * g), *(const v8bf*)(brow + kk + 16 + 8 * g),
                0,1,2,3,4,5,6,7,8,9,10,11,12,13,14,15);
            acc = __builtin_amdgcn_wmma_f32_16x16x32_bf16(
                false, afrag, false, bfrag, (short)0, acc, false, false);
        }
        const float cn = cnorm[col];
        #pragma unroll
        for (int r = 0; r < 8; ++r) {
            int mm = r + 8 * g;                 // C/D layout: m = r + 8*(L>>4)
            dist_s[mm * KK + col] = cn - 2.0f * acc[r];
        }
    }
    __syncthreads();

    // per-row min/argmin: 16 threads per row, 128-col chunks, first-hit ties
    const int pm  = tid >> 4;
    const int sub = tid & 15;
    {
        const float* dr = dist_s + pm * KK;
        float best = 3.0e38f; int bidx = sub * (KK / 16);
        for (int c = sub * (KK / 16); c < (sub + 1) * (KK / 16); ++c) {
            float v = dr[c];
            if (v < best) { best = v; bidx = c; }
        }
        red_val[tid] = best; red_idx[tid] = bidx;
    }
    __syncthreads();
    if (sub == 0) {
        float rb = red_val[tid]; int ri = red_idx[tid];
        for (int s = 1; s < 16; ++s) {          // ascending chunks: strict <
            float v = red_val[tid + s];         // keeps first occurrence
            if (v < rb) { rb = v; ri = red_idx[tid + s]; }
        }
        row_min[pm] = rb;
        if (idx_i) { idx_i[n0 + pm] = ri; idx_f[n0 + pm] = (float)ri; }
    }
    __syncthreads();

    // sum of exp(min - dist)  (== softmax(-dist) normalizer)
    {
        const float rm = row_min[pm];
        const float* dr = dist_s + pm * KK;
        float s = 0.0f;
        for (int c = sub * (KK / 16); c < (sub + 1) * (KK / 16); ++c)
            s += expf(rm - dr[c]);
        red_val[tid] = s;
    }
    __syncthreads();
    if (sub == 0) {
        float t = 0.0f;
        for (int s = 0; s < 16; ++s) t += red_val[tid + s];  // fixed order
        row_inv[pm] = 1.0f / t;
    }
    __syncthreads();

    // coalesced prob writes
    for (int i = tid; i < 16 * KK; i += 256) {
        int mm = i >> 11, c = i & (KK - 1);
        prob_out[(size_t)(n0 + mm) * KK + c] =
            expf(row_min[mm] - dist_s[mm * KK + c]) * row_inv[mm];
    }
}

// ---------------------------------------------------------------------------
// 4) z_q gather back to (b,d,h,w) + per-block SSE partial. One block per (b,h).
// ---------------------------------------------------------------------------
__global__ __launch_bounds__(256) void zq_loss_kernel(
    const float* __restrict__ cb, const float* __restrict__ z,
    const int* __restrict__ idx, float* __restrict__ zq_out,
    float* __restrict__ partials)
{
    const int bid = blockIdx.x;
    const int b = bid >> 5, h = bid & 31;
    const int tid = threadIdx.x;
    const int w = tid & 31, ds = tid >> 5;
    const size_t nbase = (size_t)b * (HH * WW) + h * WW;
    const int myidx = idx[nbase + w];
    float acc = 0.0f;
    for (int d = ds; d < DD; d += 8) {
        float c = cb[(size_t)myidx * DD + d];
        size_t off = (((size_t)b * DD + d) * HH + h) * WW + w;
        float zv = z[off];
        zq_out[off] = c;                  // straight-through value == codebook row
        float df = c - zv;
        acc += df * df;
    }
    __shared__ float rs[256];
    rs[tid] = acc; __syncthreads();
    for (int off = 128; off > 0; off >>= 1) {
        if (tid < off) rs[tid] += rs[tid + off];
        __syncthreads();
    }
    if (tid == 0) partials[bid] = rs[0];
}

// ---------------------------------------------------------------------------
// 5) deterministic final loss reduction: q_loss = (1+beta)*SSE/(N*D)
// ---------------------------------------------------------------------------
__global__ __launch_bounds__(256) void finalize_loss_kernel(
    const float* __restrict__ partials, float* __restrict__ out_loss)
{
    const int tid = threadIdx.x;
    __shared__ float rs[256];
    rs[tid] = partials[tid] + partials[tid + 256];
    __syncthreads();
    for (int off = 128; off > 0; off >>= 1) {
        if (tid < off) rs[tid] += rs[tid + off];
        __syncthreads();
    }
    if (tid == 0)
        out_loss[0] = rs[0] * (1.0f + BETA) / ((float)NN * (float)DD);
}

// ---------------------------------------------------------------------------
extern "C" void kernel_launch(void* const* d_in, const int* in_sizes, int n_in,
                              void* d_out, int out_size, void* d_ws, size_t ws_size,
                              hipStream_t stream) {
    (void)in_sizes; (void)n_in; (void)out_size; (void)ws_size;
    const float* z    = (const float*)d_in[0];
    const float* zpos = (const float*)d_in[1];
    const float* cb   = (const float*)d_in[2];
    float* out = (float*)d_out;

    char* ws = (char*)d_ws;
    bf16*  cb16    = (bf16*)(ws);                         //  2,097,152 B
    float* cnorm   = (float*)(ws + 2097152);              //      8,192 B
    bf16*  A_z     = (bf16*)(ws + 2105344);               // 16,777,216 B
    bf16*  A_pos   = (bf16*)(ws + 18882560);              // 16,777,216 B
    int*   idx     = (int*)(ws + 35659776);               //     65,536 B
    float* partial = (float*)(ws + 35725312);             //      2,048 B

    float* zq_out    = out + ZQ_OFF;
    float* loss_out  = out + LOSS_OFF;
    float* prob_out  = out + PROB_OFF;
    float* pprob_out = out + PPROB_OFF;
    float* idxf_out  = out + IDX_OFF;

    prep_codebook_kernel<<<KK, 256, 0, stream>>>(cb, cb16, cnorm);
    prep_z_kernel<<<BB * HH, 256, 0, stream>>>(z, A_z);
    prep_z_kernel<<<BB * HH, 256, 0, stream>>>(zpos, A_pos);

    vq_main_kernel<<<NN / 16, 256, 0, stream>>>(A_z, cb16, cnorm, prob_out,
                                                idx, idxf_out);
    vq_main_kernel<<<NN / 16, 256, 0, stream>>>(A_pos, cb16, cnorm, pprob_out,
                                                nullptr, nullptr);

    zq_loss_kernel<<<BB * HH, 256, 0, stream>>>(cb, z, idx, zq_out, partial);
    finalize_loss_kernel<<<1, 256, 0, stream>>>(partial, loss_out);
}
